// graph_cp_pooling_21105469293087
// MI455X (gfx1250) — matrix operand order repaired and checked by reading, not compiled, not run on hardware
//
#include <hip/hip_runtime.h>
#include <hip/hip_bf16.h>

// CDNA5 / gfx1250: wave32, WMMA 16x16x32 bf16 (f32 accumulate).
typedef __attribute__((ext_vector_type(16))) __bf16 v16bf;
typedef __attribute__((ext_vector_type(8)))  __bf16 v8bf;
typedef __attribute__((ext_vector_type(4)))  __bf16 v4bf;
typedef __attribute__((ext_vector_type(8)))  float  v8f;

#define FDIM      128
#define TILE_ROWS 64
#define XS_STRIDE 136   // bf16 elems: 272B rows -> 16B aligned, 4-bank skew
#define NBLOCKS   1024  // each block loops over B/NBLOCKS segments, W staged once

// segment_ids sorted -> each segment is a contiguous row range (binary search).
// Block stages W once (WMMA B-fragment order), then loops its segments:
// GEMM(tanh) + log/sign product-reduce + x segment-sum fully in-block.
// Deterministic: no global atomics, fixed reduction trees.
__global__ __launch_bounds__(256)
void seg_gemm_prod_kernel(const float* __restrict__ x,
                          const float* __restrict__ w,
                          const int*   __restrict__ seg,
                          float* __restrict__ out,
                          int N, int B)
{
    // W pre-swizzled into WMMA B-fragment order:
    // Wf[(kstep*8 + subtile)*32 + lane][e0..15] -> 32 contiguous bytes/lane.
    __shared__ __bf16 Wf[4 * 8 * 32 * 16];          // 32 KB
    __shared__ __bf16 Xs[TILE_ROWS * XS_STRIDE];    // 17 KB, row-major
    __shared__ float  sLog[8][FDIM];                // per-wave sum log|t|
    __shared__ float  sNeg[8][FDIM];                // per-wave neg counts
    __shared__ float  sSumA[256 * 4];               // x col-sum partials
    __shared__ float  sBias[FDIM];                  // w[128,:]

    const int tid      = threadIdx.x;
    const int lane     = tid & 31;
    const int wave     = tid >> 5;
    const int half     = lane >> 4;
    const int l16      = lane & 15;
    const int rowChunk = wave >> 1;    // 4 chunks x 16 rows
    const int colHalf  = wave & 1;     // 2 halves x 64 cols

    // --- stage W -> LDS in B-fragment layout, once per block ---
    for (int f = tid; f < 128 * 128; f += 256) {
        int K = f >> 7, col = f & 127;                 // w is [K][col] row-major
        int kstep = K >> 5, halfk = (K >> 4) & 1, e = K & 15;
        int n = col >> 4, l = col & 15;
        int ln = halfk * 16 + l;                       // target lane
        Wf[((kstep * 8 + n) * 32 + ln) * 16 + e] = (__bf16)w[f];
    }
    if (tid < FDIM) sBias[tid] = w[128 * 128 + tid];

    // Each thread owns 4 consecutive x-columns for the fp32 segment-sum.
    const int cQ     = (tid & 31) * 4;
    const int rowSet = tid >> 5;

    for (int s = blockIdx.x; s < B; s += NBLOCKS) {
        __syncthreads();   // previous segment's reducer reads done; W visible
        for (int i = tid; i < 8 * FDIM; i += 256) {
            (&sLog[0][0])[i] = 0.f;
            (&sNeg[0][0])[i] = 0.f;
        }
        float s0 = 0.f, s1 = 0.f, s2 = 0.f, s3 = 0.f;

        // --- binary search this segment's [rowStart, rowEnd) ---
        int lo = 0, hi = N;
        while (lo < hi) { int mid = (lo + hi) >> 1; if (seg[mid] <  s) lo = mid + 1; else hi = mid; }
        const int rowStart = lo;
        hi = N;
        while (lo < hi) { int mid = (lo + hi) >> 1; if (seg[mid] <= s) lo = mid + 1; else hi = mid; }
        const int rowEnd = lo;
        const int cnt    = rowEnd - rowStart;

        const int numPass = (cnt + TILE_ROWS - 1) / TILE_ROWS;
        for (int pass = 0; pass < numPass; ++pass) {
            const int passBase = pass * TILE_ROWS;
            __syncthreads();             // Xs reusable; zero-init visible
            // --- stream x tile: float4 loads, fuse col-sums, pack bf16 ---
            #pragma unroll
            for (int i = 0; i < 8; ++i) {
                int r    = rowSet + 8 * i;
                int grow = rowStart + passBase + r;
                float4 v = make_float4(0.f, 0.f, 0.f, 0.f);
                if (grow < rowEnd) v = *(const float4*)(x + grow * FDIM + cQ);
                int gnext = grow + TILE_ROWS;          // hide next pass's HBM latency
                if (gnext < rowEnd)
                    __builtin_prefetch(x + gnext * FDIM + cQ, 0, 1);
                s0 += v.x; s1 += v.y; s2 += v.z; s3 += v.w;
                v4bf p;
                p[0] = (__bf16)v.x; p[1] = (__bf16)v.y;
                p[2] = (__bf16)v.z; p[3] = (__bf16)v.w;
                *(v4bf*)(&Xs[r * XS_STRIDE + cQ]) = p;
            }
            __syncthreads();

            // --- wave: 16 rows x 64 cols = 4 ksteps x 4 subtiles of WMMA ---
            v8f zero8 = {};
            v8f acc[4] = {zero8, zero8, zero8, zero8};
            const __bf16* xrow = &Xs[(rowChunk * 16 + l16) * XS_STRIDE];
            #pragma unroll
            for (int kstep = 0; kstep < 4; ++kstep) {
                // A frag: two aligned b128 LDS loads (contiguous K = 16*half + 0..15), reg concat
                v8bf alo = *(const v8bf*)(xrow + kstep * 32 + half * 16);
                v8bf ahi = *(const v8bf*)(xrow + kstep * 32 + half * 16 + 8);
                v16bf afrag = __builtin_shufflevector(alo, ahi,
                    0,1,2,3,4,5,6,7,8,9,10,11,12,13,14,15);
                #pragma unroll
                for (int n = 0; n < 4; ++n) {
                    const __bf16* bp =
                        &Wf[((kstep * 8 + colHalf * 4 + n) * 32 + lane) * 16];
                    v8bf blo = *(const v8bf*)bp;
                    v8bf bhi = *(const v8bf*)(bp + 8);
                    v16bf bfrag = __builtin_shufflevector(blo, bhi,
                        0,1,2,3,4,5,6,7,8,9,10,11,12,13,14,15);
                    acc[n] = __builtin_amdgcn_wmma_f32_16x16x32_bf16(
                        false, afrag, false, bfrag, (short)0, acc[n],
                        false, false);
                }
            }

            // --- log|tanh(u)| = log((1-e^{-2|u|})/(1+e^{-2|u|})), sign(u) ---
            #pragma unroll
            for (int n = 0; n < 4; ++n) {
                const int col = colHalf * 64 + n * 16 + l16;
                float lsum = 0.f, nneg = 0.f;
                #pragma unroll
                for (int v = 0; v < 8; ++v) {
                    // C layout: VGPR v -> row v + 8*half, lane%16 -> col
                    int rowInSeg = passBase + rowChunk * 16 + v + 8 * half;
                    float u  = acc[n][v] + sBias[col];
                    float z  = __expf(-2.f * fabsf(u));
                    float lt = __logf(__fdividef(1.f - z, 1.f + z));
                    bool valid = rowInSeg < cnt;
                    lsum += valid ? lt : 0.f;
                    nneg += (valid && (u < 0.f)) ? 1.f : 0.f;
                }
                lsum += __shfl_xor(lsum, 16, 32);  // fold other 8 rows, same col
                nneg += __shfl_xor(nneg, 16, 32);
                if (half == 0) {                   // per-wave slice: race-free
                    sLog[wave][col] += lsum;
                    sNeg[wave][col] += nneg;
                }
            }
        }

        sSumA[tid * 4 + 0] = s0;  sSumA[tid * 4 + 1] = s1;
        sSumA[tid * 4 + 2] = s2;  sSumA[tid * 4 + 3] = s3;
        __syncthreads();
        if (tid < FDIM) {
            const int cq = tid >> 2, comp = tid & 3;
            float ss = 0.f;
            #pragma unroll
            for (int j = 0; j < 8; ++j)            // 8 rowSet owners per column
                ss += sSumA[(j * 32 + cq) * 4 + comp];
            float lsum = 0.f, nneg = 0.f;
            #pragma unroll
            for (int wv = 0; wv < 8; ++wv) { lsum += sLog[wv][tid]; nneg += sNeg[wv][tid]; }
            float sgn = ((int)nneg & 1) ? -1.f : 1.f;
            out[s * FDIM + tid] = sgn * __expf(lsum) + ss;  // empty segment -> 1.0
        }
    }
}

extern "C" void kernel_launch(void* const* d_in, const int* in_sizes, int n_in,
                              void* d_out, int out_size, void* d_ws, size_t ws_size,
                              hipStream_t stream) {
    const float* x   = (const float*)d_in[0];   // [N, 128] f32
    const float* w   = (const float*)d_in[1];   // [129, 128] f32
    const int*   seg = (const int*)d_in[2];     // [N] sorted segment ids
    float*       out = (float*)d_out;           // [4096, 128] f32

    const int N = in_sizes[2];
    const int B = out_size / FDIM;              // 4096 segments

    seg_gemm_prod_kernel<<<NBLOCKS, 256, 0, stream>>>(x, w, seg, out, N, B);
}
